// ST_BLOCK_1_3556232921194
// MI455X (gfx1250) — matrix operand order repaired and checked by reading, not compiled, not run on hardware
//
#include <hip/hip_runtime.h>
#include <hip/hip_bf16.h>
#include <math.h>

// ST_BLOCK for MI455X (gfx1250). Heavy node-dim GEMMs (adj^T*X, M^T*H/deg)
// run on v_wmma_f32_16x16x32_f16 (f16 in, f32 accumulate). Scratch ~178 MB.

#define Bdim 8
#define Cout 64
#define Ndim 512
#define Tdim 24
#define EPSF 1e-5f

typedef _Float16 v16h __attribute__((ext_vector_type(16)));
typedef _Float16 v8h  __attribute__((ext_vector_type(8)));
typedef float    v8f  __attribute__((ext_vector_type(8)));

__device__ __forceinline__ float leakyf(float x) { return x > 0.f ? x : 0.01f * x; }
__device__ __forceinline__ float sigmf(float x)  { return 1.f / (1.f + __expf(-x)); }

// ---------------- zero small stats ----------------
__global__ void k_zero(float* p, int n) {
  int i = blockIdx.x * blockDim.x + threadIdx.x;
  if (i < n) p[i] = 0.f;
}

// ---------------- stage A: pointwise conv + temporal conv ----------------
__global__ void k_stageA(const float* __restrict__ x, const float* __restrict__ w1,
                         const float* __restrict__ b1, const float* __restrict__ tw,
                         const float* __restrict__ tb, float* __restrict__ xin,
                         float* __restrict__ y) {
  int idx = blockIdx.x * blockDim.x + threadIdx.x;
  if (idx >= Bdim * Cout * Ndim * Tdim) return;
  int t = idx % Tdim;
  int n = (idx / Tdim) % Ndim;
  int o = (idx / (Tdim * Ndim)) % Cout;
  int b = idx / (Tdim * Ndim * Cout);
  const float* xb = x + (size_t)b * 32 * Ndim * Tdim + (size_t)n * Tdim + t;
  float acc_in = b1[o], acc_y = tb[o];
  #pragma unroll 4
  for (int c = 0; c < 32; ++c) {
    float xv = xb[(size_t)c * Ndim * Tdim];
    float xm = (t > 0)        ? xb[(size_t)c * Ndim * Tdim - 1] : 0.f;
    float xp = (t < Tdim - 1) ? xb[(size_t)c * Ndim * Tdim + 1] : 0.f;
    acc_in += xv * w1[o * 32 + c];
    const float* w = tw + (size_t)(o * 32 + c) * 3;
    acc_y += xm * w[0] + xv * w[1] + xp * w[2];
  }
  xin[idx] = acc_in;
  y[idx] = acc_y;
}

// ---------------- per-batch sum / sumsq reduction ----------------
__global__ void k_reduce(const float* __restrict__ src, float* __restrict__ stats,
                         int perBatch) {
  int b = blockIdx.x;
  int tid = threadIdx.x;
  const float* p = src + (size_t)b * perBatch;
  float s = 0.f, ss = 0.f;
  int stride = gridDim.y * blockDim.x;
  for (int i = blockIdx.y * blockDim.x + tid; i < perBatch; i += stride) {
    float v = p[i]; s += v; ss += v * v;
  }
  __shared__ float s1[256], s2[256];
  s1[tid] = s; s2[tid] = ss;
  __syncthreads();
  for (int o = 128; o > 0; o >>= 1) {
    if (tid < o) { s1[tid] += s1[tid + o]; s2[tid] += s2[tid + o]; }
    __syncthreads();
  }
  if (tid == 0) {
    atomicAdd(&stats[b * 2 + 0], s1[0]);
    atomicAdd(&stats[b * 2 + 1], s2[0]);
  }
}

// ---------------- layernorm (+ optional leaky) ----------------
__global__ void k_ln(const float* __restrict__ src, const float* __restrict__ stats,
                     const float* __restrict__ g, const float* __restrict__ bb,
                     float* __restrict__ dst, int applyLeaky) {
  int idx = blockIdx.x * blockDim.x + threadIdx.x;
  const int PB = Cout * Ndim * Tdim;
  if (idx >= Bdim * PB) return;
  int b = idx / PB, j = idx % PB;
  float cnt = (float)PB;
  float m = stats[b * 2] / cnt;
  float v = stats[b * 2 + 1] / cnt - m * m;
  float z = (src[idx] - m) * rsqrtf(v + EPSF) * g[j] + bb[j];
  dst[idx] = applyLeaky ? leakyf(z) : z;
}

// ---------------- adj^T / mask^T to f16 (k-contiguous rows) ----------------
__global__ void k_prepW(const float* __restrict__ adj, _Float16* __restrict__ adjT,
                        _Float16* __restrict__ MT) {
  int idx = blockIdx.x * blockDim.x + threadIdx.x;
  if (idx >= Ndim * Ndim) return;
  int q = idx / Ndim, n = idx % Ndim;
  float a = adj[n * Ndim + q];
  adjT[idx] = (_Float16)a;
  MT[idx] = (a != 0.f) ? (_Float16)1.f : (_Float16)0.f;
}

__global__ void k_deg(const float* __restrict__ adj, float* __restrict__ deginv) {
  int v = blockIdx.x * blockDim.x + threadIdx.x;
  if (v >= Ndim) return;
  float s = 0.f;
  for (int u = 0; u < Ndim; ++u) s += (adj[u * Ndim + v] != 0.f) ? 1.f : 0.f;
  deginv[v] = 1.f / fmaxf(s, 1.f);
}

// ---------------- x1 -> f16 column-major (per-col 512 contiguous n) --------
__global__ void k_convX(const float* __restrict__ x1, _Float16* __restrict__ Hc) {
  int idx = blockIdx.x * blockDim.x + threadIdx.x;
  if (idx >= Bdim * Cout * Ndim * Tdim) return;
  int l = idx % Tdim;
  int n = (idx / Tdim) % Ndim;
  int bc = idx / (Tdim * Ndim);          // b*64+c
  int col = bc * Tdim + l;
  Hc[(size_t)col * Ndim + n] = (_Float16)x1[idx];
}

// ---------------- h -> f16 column-major ----------------
__global__ void k_convH(const float* __restrict__ h, _Float16* __restrict__ Hc) {
  int idx = blockIdx.x * blockDim.x + threadIdx.x;
  if (idx >= Bdim * Cout * 2 * Ndim * Tdim) return;
  int l = idx % Tdim;
  int n = (idx / Tdim) % Ndim;
  int bck = idx / (Tdim * Ndim);         // (b*64+c)*2+k
  int col = bck * Tdim + l;
  Hc[(size_t)col * Ndim + n] = (_Float16)h[idx];
}

// ---------------- WMMA GEMM: out[q,col] = sum_n At[q,n]*Bc[col,n] ----------
// At: [512 x 512] f16, k-contiguous rows.  Bc: [COLS x 512] f16.
// out offset = ((col/24)*mul + add)*12288 + (col%24) + q*24; row scale optional.
__global__ __launch_bounds__(256)
void k_gemm_wmma(const _Float16* __restrict__ At, const _Float16* __restrict__ Bc,
                 float* __restrict__ out, const float* __restrict__ rowscale,
                 int COLS, int mul, int add) {
  const int lane = threadIdx.x;                 // 0..31
  const int wave = threadIdx.y;                 // 0..7
  const int q0   = blockIdx.y * 16;
  const int col0 = (blockIdx.x * 8 + wave) * 16;
  const int m    = lane & 15;
  const int half = lane >> 4;
  const _Float16* aRow = At + (size_t)(q0 + m) * Ndim;
  const _Float16* bRow = Bc + (size_t)(col0 + m) * Ndim;
  v8f acc = {};
  #pragma unroll 4
  for (int kb = 0; kb < Ndim; kb += 32) {
    if (kb + 32 < Ndim) {
      __builtin_prefetch(aRow + kb + 32, 0, 0);
      __builtin_prefetch(bRow + kb + 32, 0, 0);
    }
    union { v16h v; v8h h[2]; } a, b;
    // A 16x32 f16 layout: lane half owns K j+8h (j<8) and 16+(j-8)+8h (j>=8)
    a.h[0] = *(const v8h*)(aRow + kb + 8 * half);
    a.h[1] = *(const v8h*)(aRow + kb + 16 + 8 * half);
    // B 32x16 f16 layout: same K order as A: K j+8h (j<8) and 16+(j-8)+8h (j>=8)
    b.h[0] = *(const v8h*)(bRow + kb + 8 * half);
    b.h[1] = *(const v8h*)(bRow + kb + 16 + 8 * half);
    acc = __builtin_amdgcn_wmma_f32_16x16x32_f16(
        false, a.v, false, b.v, (short)0, acc, false, false);
  }
  int col = col0 + m;
  size_t base = ((size_t)(col / Tdim) * (size_t)mul + (size_t)add) * (size_t)(Ndim * Tdim)
              + (size_t)(col % Tdim);
  #pragma unroll
  for (int r = 0; r < 8; ++r) {
    int q = q0 + r + 8 * half;                  // C/D: VGPR r -> M=r (+8 for hi half)
    float v = acc[r];
    if (rowscale) v *= rowscale[q];
    out[base + (size_t)q * Tdim] = v;
  }
}

// ---------------- h[:, :, k=0] = 4*x1 ----------------
__global__ void k_hk0(const float* __restrict__ x1, float* __restrict__ h) {
  int idx = blockIdx.x * blockDim.x + threadIdx.x;
  if (idx >= Bdim * Cout * Ndim * Tdim) return;
  int l = idx % Tdim;
  int n = (idx / Tdim) % Ndim;
  int bc = idx / (Tdim * Ndim);
  h[((size_t)bc * 2 + 0) * (Ndim * Tdim) + (size_t)n * Tdim + l] = 4.f * x1[idx];
}

// ---------------- SAGE combine (in place over h) ----------------
__global__ void k_sage(float* __restrict__ h, const float* __restrict__ g,
                       const float* __restrict__ ws, const float* __restrict__ wn,
                       const float* __restrict__ bias) {
  __shared__ float sws[576], swn[576], sb[24];
  int tid = threadIdx.x;
  for (int i = tid; i < 576; i += blockDim.x) { sws[i] = ws[i]; swn[i] = wn[i]; }
  if (tid < 24) sb[tid] = bias[tid];
  __syncthreads();
  int idx = blockIdx.x * blockDim.x + tid;    // (b,c,k,n) row index
  if (idx >= Bdim * Cout * 2 * Ndim) return;
  float hr[Tdim], gr[Tdim], outr[Tdim];
  float* hp = h + (size_t)idx * Tdim;
  const float* gp = g + (size_t)idx * Tdim;
  #pragma unroll
  for (int l = 0; l < Tdim; ++l) { hr[l] = hp[l]; gr[l] = gp[l]; }
  for (int mo = 0; mo < Tdim; ++mo) {
    float s = sb[mo];
    #pragma unroll
    for (int l = 0; l < Tdim; ++l)
      s += hr[l] * sws[mo * Tdim + l] + gr[l] * swn[mo * Tdim + l];
    outr[mo] = s;
  }
  #pragma unroll
  for (int mo = 0; mo < Tdim; ++mo) hp[mo] = outr[mo];
}

// ---------------- gating: sigmoid(gate)*leaky(filt) ----------------
__global__ void k_gate(const float* __restrict__ h, float* __restrict__ x1) {
  int idx = blockIdx.x * blockDim.x + threadIdx.x;
  if (idx >= Bdim * Cout * Ndim * Tdim) return;
  int t = idx % Tdim;
  int n = (idx / Tdim) % Ndim;
  int cc = (idx / (Tdim * Ndim)) % Cout;
  int b  = idx / (Tdim * Ndim * Cout);
  size_t tail = (size_t)n * Tdim + t;
  size_t fOff = ((size_t)(b * Cout + (cc >> 1)) * 2 + (cc & 1)) * (Ndim * Tdim) + tail;
  size_t gOff = ((size_t)(b * Cout + 32 + (cc >> 1)) * 2 + (cc & 1)) * (Ndim * Tdim) + tail;
  x1[idx] = sigmf(h[gOff]) * leakyf(h[fOff]);
}

// ---------------- f1 = sum_c x1*t_w1 ; f2 = sum_n x1*t_w2 ----------------
__global__ void k_f1(const float* __restrict__ x1, const float* __restrict__ w,
                     float* __restrict__ f1) {
  int idx = blockIdx.x * blockDim.x + threadIdx.x;
  if (idx >= Bdim * Tdim * Ndim) return;
  int n = idx % Ndim;
  int l = (idx / Ndim) % Tdim;
  int b = idx / (Ndim * Tdim);
  float s = 0.f;
  const float* p = x1 + ((size_t)b * Cout * Ndim + n) * Tdim + l;
  for (int c = 0; c < Cout; ++c) s += p[(size_t)c * Ndim * Tdim] * w[c];
  f1[idx] = s;
}

__global__ void k_f1conv(const float* __restrict__ f1, const float* __restrict__ wd,
                         float* __restrict__ f1c) {
  int idx = blockIdx.x * blockDim.x + threadIdx.x;
  if (idx >= Bdim * Tdim * Ndim) return;
  int n = idx % Ndim;
  int o = (idx / Ndim) % Tdim;
  int b = idx / (Ndim * Tdim);
  float s = 0.f;
  for (int i = 0; i < Tdim; ++i) {
    float xm = (n > 0)        ? f1[((size_t)b * Tdim + i) * Ndim + n - 1] : 0.f;
    float xp = (n < Ndim - 1) ? f1[((size_t)b * Tdim + i) * Ndim + n + 1] : 0.f;
    s += wd[(o * Tdim + i) * 2 + 0] * xm + wd[(o * Tdim + i) * 2 + 1] * xp;
  }
  f1c[idx] = s;
}

__global__ void k_f2(const float* __restrict__ x1, const float* __restrict__ w,
                     float* __restrict__ f2) {
  int idx = blockIdx.x * blockDim.x + threadIdx.x;
  if (idx >= Bdim * Cout * Tdim) return;
  int l = idx % Tdim;
  int bc = idx / Tdim;                       // b*64+c
  float s = 0.f;
  const float* p = x1 + (size_t)bc * Ndim * Tdim + l;
  for (int n = 0; n < Ndim; ++n) s += p[(size_t)n * Tdim] * w[n];
  f2[idx] = s;
}

__global__ void k_f2conv(const float* __restrict__ f2, const float* __restrict__ wd,
                         float* __restrict__ f2c) {
  int idx = blockIdx.x * blockDim.x + threadIdx.x;
  if (idx >= Bdim * Cout * Tdim) return;
  int t = idx % Tdim;
  int o = (idx / Tdim) % Cout;
  int b = idx / (Tdim * Cout);
  float s = 0.f;
  for (int i = 0; i < Cout; ++i) {
    float xm = (t > 0)        ? f2[((size_t)b * Cout + i) * Tdim + t - 1] : 0.f;
    float xp = (t < Tdim - 1) ? f2[((size_t)b * Cout + i) * Tdim + t + 1] : 0.f;
    s += wd[(o * Cout + i) * 2 + 0] * xm + wd[(o * Cout + i) * 2 + 1] * xp;
  }
  f2c[idx] = s;
}

// tmp[b,l,c] = sum_n f1c[b,l,n]*t_W[n,c]
__global__ void k_fW(const float* __restrict__ f1c, const float* __restrict__ tW,
                     float* __restrict__ tmp) {
  int idx = blockIdx.x * blockDim.x + threadIdx.x;
  if (idx >= Bdim * Tdim * Cout) return;
  int c = idx % Cout;
  int bl = idx / Cout;
  float s = 0.f;
  const float* p = f1c + (size_t)bl * Ndim;
  for (int n = 0; n < Ndim; ++n) s += p[n] * tW[n * Cout + c];
  tmp[idx] = s;
}

// sig[b,l,q] = sigmoid(sum_c tmp[b,l,c]*f2c[b,c,q] + t_b[l,q])
__global__ void k_siglog(const float* __restrict__ tmp, const float* __restrict__ f2c,
                         const float* __restrict__ tb, float* __restrict__ sig) {
  int idx = blockIdx.x * blockDim.x + threadIdx.x;
  if (idx >= Bdim * Tdim * Tdim) return;
  int q = idx % Tdim;
  int l = (idx / Tdim) % Tdim;
  int b = idx / (Tdim * Tdim);
  float s = tb[l * Tdim + q];
  for (int c = 0; c < Cout; ++c)
    s += tmp[((size_t)b * Tdim + l) * Cout + c] * f2c[((size_t)b * Cout + c) * Tdim + q];
  sig[idx] = sigmf(s);
}

// log2[b,l,q] = sum_k t_v[l,k]*sig[b,k,q]
__global__ void k_tv(const float* __restrict__ sig, const float* __restrict__ tv,
                     float* __restrict__ lg) {
  int idx = blockIdx.x * blockDim.x + threadIdx.x;
  if (idx >= Bdim * Tdim * Tdim) return;
  int q = idx % Tdim;
  int l = (idx / Tdim) % Tdim;
  int b = idx / (Tdim * Tdim);
  float s = 0.f;
  for (int k = 0; k < Tdim; ++k)
    s += tv[l * Tdim + k] * sig[((size_t)b * Tdim + k) * Tdim + q];
  lg[idx] = s;
}

// BN stats over (b, l) per q
__global__ void k_bnstats(const float* __restrict__ lg, float* __restrict__ bn) {
  int q = blockIdx.x * blockDim.x + threadIdx.x;
  if (q >= Tdim) return;
  float s = 0.f, ss = 0.f;
  for (int b = 0; b < Bdim; ++b)
    for (int l = 0; l < Tdim; ++l) {
      float v = lg[((size_t)b * Tdim + l) * Tdim + q];
      s += v; ss += v * v;
    }
  float cnt = (float)(Bdim * Tdim);
  float m = s / cnt;
  bn[q * 2 + 0] = m;
  bn[q * 2 + 1] = ss / cnt - m * m;
}

// BN + block-masked softmax over q -> coefs[b,l,q]
__global__ void k_softmax(const float* __restrict__ lg, const float* __restrict__ bn,
                          const float* __restrict__ bng, const float* __restrict__ bnb,
                          float* __restrict__ coef) {
  int idx = blockIdx.x * blockDim.x + threadIdx.x;
  if (idx >= Bdim * Tdim) return;
  int l = idx % Tdim, b = idx / Tdim;
  float z[Tdim];
  float mx = -1e30f;
  for (int q = 0; q < Tdim; ++q) {
    float v = (lg[((size_t)b * Tdim + l) * Tdim + q] - bn[q * 2]) *
                  rsqrtf(bn[q * 2 + 1] + EPSF) * bng[q] + bnb[q];
    bool allowed = ((l < 12) == (q < 12));
    z[q] = allowed ? v : -1e30f;
    if (z[q] > mx) mx = z[q];
  }
  float sum = 0.f;
  for (int q = 0; q < Tdim; ++q) { z[q] = (z[q] > -1e29f) ? __expf(z[q] - mx) : 0.f; sum += z[q]; }
  float inv = 1.f / sum;
  for (int q = 0; q < Tdim; ++q) coef[((size_t)b * Tdim + l) * Tdim + q] = z[q] * inv;
}

// T_coef output = coefs transposed per batch
__global__ void k_tcoef(const float* __restrict__ coef, float* __restrict__ outT) {
  int idx = blockIdx.x * blockDim.x + threadIdx.x;
  if (idx >= Bdim * Tdim * Tdim) return;
  int j = idx % Tdim;
  int i = (idx / Tdim) % Tdim;
  int b = idx / (Tdim * Tdim);
  outT[idx] = coef[((size_t)b * Tdim + j) * Tdim + i];
}

// x2 = x1 @ T_coef ; w = leaky(x2) + x_input
__global__ void k_apply(const float* __restrict__ x1, const float* __restrict__ coef,
                        const float* __restrict__ xin, float* __restrict__ w) {
  int idx = blockIdx.x * blockDim.x + threadIdx.x;
  if (idx >= Bdim * Cout * Ndim * Tdim) return;
  int q = idx % Tdim;
  int bcn = idx / Tdim;
  int b = bcn / (Cout * Ndim);
  const float* xr = x1 + (size_t)bcn * Tdim;
  const float* cr = coef + ((size_t)b * Tdim + q) * Tdim;   // coefs[b,q,l]
  float s = 0.f;
  #pragma unroll
  for (int l = 0; l < Tdim; ++l) s += xr[l] * cr[l];
  w[idx] = leakyf(s) + xin[idx];
}

__global__ void k_out(const float* __restrict__ w, const float* __restrict__ stats,
                      const float* __restrict__ g, const float* __restrict__ bb,
                      float* __restrict__ out) {
  int idx = blockIdx.x * blockDim.x + threadIdx.x;
  const int PB = Cout * Ndim * Tdim;
  if (idx >= Bdim * PB) return;
  int b = idx / PB, j = idx % PB;
  float cnt = (float)PB;
  float m = stats[b * 2] / cnt;
  float v = stats[b * 2 + 1] / cnt - m * m;
  out[idx] = (w[idx] - m) * rsqrtf(v + EPSF) * g[j] + bb[j];
}

extern "C" void kernel_launch(void* const* d_in, const int* in_sizes, int n_in,
                              void* d_out, int out_size, void* d_ws, size_t ws_size,
                              hipStream_t stream) {
  (void)in_sizes; (void)n_in; (void)out_size; (void)ws_size;
  const float* x      = (const float*)d_in[0];
  const float* adj    = (const float*)d_in[1];
  const float* conv1w = (const float*)d_in[2];
  const float* conv1b = (const float*)d_in[3];
  const float* tcw    = (const float*)d_in[4];
  const float* tcb    = (const float*)d_in[5];
  const float* lng    = (const float*)d_in[6];
  const float* lnb    = (const float*)d_in[7];
  const float* s0ws   = (const float*)d_in[8];
  const float* s0wn   = (const float*)d_in[9];
  const float* s0b    = (const float*)d_in[10];
  const float* s1ws   = (const float*)d_in[11];
  const float* s1wn   = (const float*)d_in[12];
  const float* s1b    = (const float*)d_in[13];
  const float* tw1    = (const float*)d_in[14];
  const float* tw2    = (const float*)d_in[15];
  const float* twd1   = (const float*)d_in[16];
  const float* twd2   = (const float*)d_in[17];
  const float* tW     = (const float*)d_in[18];
  const float* tb     = (const float*)d_in[19];
  const float* tv     = (const float*)d_in[20];
  const float* bng    = (const float*)d_in[21];
  const float* bnb    = (const float*)d_in[22];
  float* out = (float*)d_out;

  const size_t NBCT = (size_t)Bdim * Cout * Ndim * Tdim;   // 6,291,456
  const size_t NH   = NBCT * 2;                            // 12,582,912

  // ---- workspace partition (~178 MB) ----
  float* f_xin    = (float*)d_ws;             // 6.29M
  float* f_x1     = f_xin + NBCT;             // 6.29M (y -> x1 -> gated x1)
  float* f_h      = f_x1 + NBCT;              // 12.58M
  float* f_g      = f_h + NH;                 // 12.58M
  _Float16* h_f16 = (_Float16*)(f_g + NH);    // 12.58M halves
  _Float16* adjT  = h_f16 + NH;               // 262144 halves
  _Float16* MT    = adjT + Ndim * Ndim;       // 262144 halves
  float* f_deginv = (float*)(MT + Ndim * Ndim);
  float* f_stats  = f_deginv + Ndim;          // 16
  float* f_stats2 = f_stats + 16;             // 16
  float* f_f1     = f_stats2 + 16;            // 98304
  float* f_f1c    = f_f1 + Bdim * Tdim * Ndim;
  float* f_f2     = f_f1c + Bdim * Tdim * Ndim;  // 12288
  float* f_f2c    = f_f2 + Bdim * Cout * Tdim;
  float* f_tmp    = f_f2c + Bdim * Cout * Tdim;  // 12288
  float* f_sig    = f_tmp + Bdim * Tdim * Cout;  // 4608
  float* f_lg     = f_sig + Bdim * Tdim * Tdim;
  float* f_bn     = f_lg + Bdim * Tdim * Tdim;   // 48
  float* f_coef   = f_bn + 48;                   // 4608

  const int TPB = 256;
  const int GB  = (int)((NBCT + TPB - 1) / TPB);     // 24576
  const int GH  = (int)((NH + TPB - 1) / TPB);       // 49152

  k_zero<<<1, 64, 0, stream>>>(f_stats, 32);
  k_stageA<<<GB, TPB, 0, stream>>>(x, conv1w, conv1b, tcw, tcb, f_xin, f_x1);
  k_reduce<<<dim3(Bdim, 96), TPB, 0, stream>>>(f_x1, f_stats, Cout * Ndim * Tdim);
  k_ln<<<GB, TPB, 0, stream>>>(f_x1, f_stats, lng, lnb, f_x1, 1);

  k_prepW<<<(Ndim * Ndim) / TPB, TPB, 0, stream>>>(adj, adjT, MT);
  k_deg<<<2, TPB, 0, stream>>>(adj, f_deginv);

  // h[k=1] = adj^T @ x1  (WMMA),  h[k=0] = 4*x1
  k_convX<<<GB, TPB, 0, stream>>>(f_x1, h_f16);
  k_gemm_wmma<<<dim3(96, 32), dim3(32, 8), 0, stream>>>(
      adjT, h_f16, f_h, nullptr, Bdim * Cout * Tdim, 2, 1);
  k_hk0<<<GB, TPB, 0, stream>>>(f_x1, f_h);

  // two SAGE layers: g = (M^T @ h)/deg (WMMA), then h = h@ws^T + g@wn^T + b
  for (int s = 0; s < 2; ++s) {
    k_convH<<<GH, TPB, 0, stream>>>(f_h, h_f16);
    k_gemm_wmma<<<dim3(192, 32), dim3(32, 8), 0, stream>>>(
        MT, h_f16, f_g, f_deginv, Bdim * Cout * 2 * Tdim, 1, 0);
    k_sage<<<(int)((Bdim * Cout * 2 * Ndim) / TPB), TPB, 0, stream>>>(
        f_h, f_g, s ? s1ws : s0ws, s ? s1wn : s0wn, s ? s1b : s0b);
  }

  k_gate<<<GB, TPB, 0, stream>>>(f_h, f_x1);

  // tiny attention head
  k_f1<<<(Bdim * Tdim * Ndim) / TPB, TPB, 0, stream>>>(f_x1, tw1, f_f1);
  k_f1conv<<<(Bdim * Tdim * Ndim) / TPB, TPB, 0, stream>>>(f_f1, twd1, f_f1c);
  k_f2<<<(Bdim * Cout * Tdim) / TPB, TPB, 0, stream>>>(f_x1, tw2, f_f2);
  k_f2conv<<<(Bdim * Cout * Tdim) / TPB, TPB, 0, stream>>>(f_f2, twd2, f_f2c);
  k_fW<<<(Bdim * Tdim * Cout) / TPB, TPB, 0, stream>>>(f_f1c, tW, f_tmp);
  k_siglog<<<18, TPB, 0, stream>>>(f_tmp, f_f2c, tb, f_sig);
  k_tv<<<18, TPB, 0, stream>>>(f_sig, tv, f_lg);
  k_bnstats<<<1, 32, 0, stream>>>(f_lg, f_bn);
  k_softmax<<<1, 192, 0, stream>>>(f_lg, f_bn, bng, bnb, f_coef);

  // outputs #2 (adj passthrough) and #3 (T_coef)
  hipMemcpyAsync(out + NBCT, adj, (size_t)Ndim * Ndim * sizeof(float),
                 hipMemcpyDeviceToDevice, stream);
  k_tcoef<<<18, TPB, 0, stream>>>(f_coef, out + NBCT + Ndim * Ndim);

  // coef apply + residual + final layernorm -> output #1
  k_apply<<<GB, TPB, 0, stream>>>(f_x1, f_coef, f_xin, f_h);
  k_reduce<<<dim3(Bdim, 96), TPB, 0, stream>>>(f_h, f_stats2, Cout * Ndim * Tdim);
  k_out<<<GB, TPB, 0, stream>>>(f_h, f_stats2, lng, lnb, out);
}
